// KNNModule_2946347565933
// MI455X (gfx1250) — matrix operand order repaired and checked by reading, not compiled, not run on hardware
//
#include <hip/hip_runtime.h>
#include <hip/hip_bf16.h>

// MI455X / gfx1250, wave32. Memory-bound pipeline: every [N,128] GEMM uses
// v_wmma_f32_16x16x32_bf16 with fused bias + BN-statistics epilogue; the
// BN(scale,shift)+LeakyReLU of each layer is fused into the bf16 A-operand
// load of the *next* GEMM, except one standalone pass (apply_bn_k) that materializes W after readin.
//
// ws layout (assumes ws_size >= ~410 MB):
//   Y  : N*128 f32  (current GEMM output / BN-pending activations)
//   W  : N*128 f32  (residual hidden state)
//   stats : 7 slots * 256 f32  (sum[128] || sumsq[128])
//   st    : 7 slots * 256 f32  (scale[128] || shift[128])
//   packed bf16 transposed weights (10 * 128x128 + 4 * 16x128)

typedef __bf16 bf16_t;
typedef __attribute__((ext_vector_type(16))) __bf16 v16bf;
typedef __attribute__((ext_vector_type(8)))  float  v8f;

static constexpr int   NROWS = 400000;   // divisible by 128
static constexpr float INV_N = 1.0f / 400000.0f;
static constexpr float EPS   = 1e-5f;
static constexpr float SLOPE = 0.01f;

__device__ __forceinline__ float bnlrelu(float v, float s, float t) {
    v = fmaf(v, s, t);
    return v >= 0.0f ? v : SLOPE * v;
}

// Build one 16x32 bf16 A fragment (contiguous k order, not the ISA table order) for k-step base c1.
// Lane holds row m = lane&15; element j -> k=c1+j (contiguous 16 k per lane half).
template<int APPLY>
__device__ __forceinline__ v16bf load_a_frag(const float* __restrict__ xrow,
                                             const float* __restrict__ st,
                                             int c1) {
    float tmp[16];
    *(float4*)(tmp + 0)  = *(const float4*)(xrow + c1);
    *(float4*)(tmp + 4)  = *(const float4*)(xrow + c1 + 4);
    *(float4*)(tmp + 8)  = *(const float4*)(xrow + c1 + 8);
    *(float4*)(tmp + 12) = *(const float4*)(xrow + c1 + 12);
    v16bf a;
#pragma unroll
    for (int j = 0; j < 16; ++j) {
        int k = c1 + j;
        float v = tmp[j];
        if (APPLY) v = bnlrelu(v, st[k], st[128 + k]);
        a[j] = (bf16_t)v;
    }
    return a;
}

// ---------------- small helper kernels ----------------

__global__ void zero_stats_k(float* __restrict__ p) {
    p[blockIdx.x * 256 + threadIdx.x] = 0.0f;
}

// Transpose+convert f32 weight [128][ncols] (cols colOff..colOff+127) ->
// bf16 Wt[n][k] (n = out channel, 128 consecutive k per row).
__global__ void pack_wt_k(const float* __restrict__ src, int ncols, int colOff,
                          bf16_t* __restrict__ dst) {
    int i = blockIdx.x * 256 + threadIdx.x;     // 0..16383
    int n = i >> 7, k = i & 127;
    dst[n * 128 + k] = (bf16_t)src[k * ncols + n + colOff];
}

// Position tile: 16x128 bf16, rows 0..1 = dW1 cols 0..1, rows 2..15 zero.
__global__ void pack_pos_k(const float* __restrict__ src, bf16_t* __restrict__ dst) {
    int i = blockIdx.x * 256 + threadIdx.x;     // 0..2047
    int n = i >> 7, k = i & 127;
    float v = (n < 2) ? src[k * 130 + n] : 0.0f;
    dst[n * 128 + k] = (bf16_t)v;
}

// sum/sumsq -> per-channel scale/shift for fused BN.
__global__ void finalize_k(const float* __restrict__ sums,
                           const float* __restrict__ g,
                           const float* __restrict__ be,
                           float* __restrict__ st) {
    int c = threadIdx.x;                        // 128 threads
    float mu  = sums[c] * INV_N;
    float var = fmaf(sums[128 + c], INV_N, -mu * mu);
    float s   = g[c] * rsqrtf(var + EPS);
    st[c]        = s;
    st[128 + c]  = fmaf(-mu, s, be[c]);
}

// readin layer 0: y = weights[N,2] @ W0[2,128] + b0, with fused stats.
__global__ __launch_bounds__(256) void readin0_k(
        const float* __restrict__ wts, const float* __restrict__ W0,
        const float* __restrict__ b0, float* __restrict__ y,
        float* __restrict__ gstats) {
    __shared__ float ls[256];
    int tid = threadIdx.x;
    ls[tid] = 0.0f;
    __syncthreads();
    long gidx = (long)blockIdx.x * 256 + tid;   // thread = (row, 4 cols)
    long row  = gidx >> 5;
    int  c0   = (int)(gidx & 31) * 4;
    float2 in = *(const float2*)(wts + row * 2);
    float4 w0 = *(const float4*)(W0 + c0);
    float4 w1 = *(const float4*)(W0 + 128 + c0);
    float4 bb = *(const float4*)(b0 + c0);
    float4 v;
    v.x = fmaf(in.x, w0.x, fmaf(in.y, w1.x, bb.x));
    v.y = fmaf(in.x, w0.y, fmaf(in.y, w1.y, bb.y));
    v.z = fmaf(in.x, w0.z, fmaf(in.y, w1.z, bb.z));
    v.w = fmaf(in.x, w0.w, fmaf(in.y, w1.w, bb.w));
    *(float4*)(y + row * 128 + c0) = v;
    atomicAdd(&ls[c0 + 0], v.x);             atomicAdd(&ls[c0 + 1], v.y);
    atomicAdd(&ls[c0 + 2], v.z);             atomicAdd(&ls[c0 + 3], v.w);
    atomicAdd(&ls[128 + c0 + 0], v.x * v.x); atomicAdd(&ls[128 + c0 + 1], v.y * v.y);
    atomicAdd(&ls[128 + c0 + 2], v.z * v.z); atomicAdd(&ls[128 + c0 + 3], v.w * v.w);
    __syncthreads();
    atomicAdd(&gstats[tid], ls[tid]);
}

// Materialize hidden state once after readin: W = lrelu(bn(Y)).
__global__ __launch_bounds__(256) void apply_bn_k(const float* __restrict__ src,
                                                  const float* __restrict__ st,
                                                  float* __restrict__ dst) {
    long i  = (long)blockIdx.x * 256 + threadIdx.x;  // one float4 each
    int  c0 = (int)(i & 31) * 4;
    float4 v  = *(const float4*)(src + i * 4);
    float4 s4 = *(const float4*)(st + c0);
    float4 t4 = *(const float4*)(st + 128 + c0);
    v.x = bnlrelu(v.x, s4.x, t4.x);
    v.y = bnlrelu(v.y, s4.y, t4.y);
    v.z = bnlrelu(v.z, s4.z, t4.z);
    v.w = bnlrelu(v.w, s4.w, t4.w);
    *(float4*)(dst + i * 4) = v;
}

// ---------------- main WMMA GEMM kernels ----------------

// y[N,128] = act(x)[N,128] @ Wt^T + bias, fused per-channel sum/sumsq.
// APPLY=1: x is a raw linear output, apply scale/shift+lrelu while loading.
// Wave = one 16-row strip; 8 col-tiles x 4 k-steps = 32 WMMAs per wave.
template<int APPLY>
__global__ __launch_bounds__(256) void gemm128_stats_k(
        const float* __restrict__ x, const bf16_t* __restrict__ wt,
        const float* __restrict__ bias, const float* __restrict__ stIn,
        float* __restrict__ y, float* __restrict__ gstats) {
    __shared__ float ls[256];
    int tid = threadIdx.x;
    ls[tid] = 0.0f;
    __syncthreads();
    int wave = tid >> 5, lane = tid & 31;
    int half = lane >> 4, nn = lane & 15;
    long m0 = (long)blockIdx.x * 128 + wave * 16;

    v16bf a[4];
    const float* xrow = x + (m0 + nn) * 128;
#pragma unroll
    for (int ks = 0; ks < 4; ++ks)
        a[ks] = load_a_frag<APPLY>(xrow, stIn, ks * 32 + half * 16);

    float* ybase = y + (m0 + half * 8) * 128 + nn;
#pragma unroll
    for (int t = 0; t < 8; ++t) {
        v8f acc = {};
#pragma unroll
        for (int ks = 0; ks < 4; ++ks) {
            v16bf b = *(const v16bf*)(wt + (t * 16 + nn) * 128 + ks * 32 + half * 16);
            acc = __builtin_amdgcn_wmma_f32_16x16x32_bf16(
                false, a[ks], false, b, (short)0, acc, false, false);
        }
        int   c  = t * 16 + nn;
        float bv = bias[c];
        float ps = 0.0f, pss = 0.0f;
        float* yp = ybase + t * 16;
#pragma unroll
        for (int r = 0; r < 8; ++r) {
            float v = acc[r] + bv;
            yp[(long)r * 128] = v;
            ps += v; pss += v * v;
        }
        atomicAdd(&ls[c], ps);
        atomicAdd(&ls[128 + c], pss);
    }
    __syncthreads();
    atomicAdd(&gstats[tid], ls[tid]);
}

// Block delta GEMM: h = lrelu(bn(yA)); w += h@Wt_w + bW; pos += h@Wt_p + bP.
__global__ __launch_bounds__(256) void gemm_delta_k(
        const float* __restrict__ yA, const bf16_t* __restrict__ wtW,
        const bf16_t* __restrict__ wtP, const float* __restrict__ biasW,
        const float* __restrict__ biasP, const float* __restrict__ stIn,
        const float* __restrict__ posIn, float* __restrict__ w,
        float* __restrict__ posOut, int first) {
    int tid = threadIdx.x;
    int wave = tid >> 5, lane = tid & 31;
    int half = lane >> 4, nn = lane & 15;
    long m0 = (long)blockIdx.x * 128 + wave * 16;

    v16bf a[4];
    const float* xrow = yA + (m0 + nn) * 128;
#pragma unroll
    for (int ks = 0; ks < 4; ++ks)
        a[ks] = load_a_frag<1>(xrow, stIn, ks * 32 + half * 16);

    float* wbase = w + (m0 + half * 8) * 128 + nn;
#pragma unroll
    for (int t = 0; t < 8; ++t) {
        v8f acc = {};
#pragma unroll
        for (int ks = 0; ks < 4; ++ks) {
            v16bf b = *(const v16bf*)(wtW + (t * 16 + nn) * 128 + ks * 32 + half * 16);
            acc = __builtin_amdgcn_wmma_f32_16x16x32_bf16(
                false, a[ks], false, b, (short)0, acc, false, false);
        }
        float bv = biasW[t * 16 + nn];
        float* wp = wbase + t * 16;
#pragma unroll
        for (int r = 0; r < 8; ++r)
            wp[(long)r * 128] += acc[r] + bv;   // unique ownership, no atomics
    }

    // position tile (cols 2..15 of Wt_p are zero) — WMMA before any divergence
    v8f pacc = {};
#pragma unroll
    for (int ks = 0; ks < 4; ++ks) {
        v16bf b = *(const v16bf*)(wtP + nn * 128 + ks * 32 + half * 16);
        pacc = __builtin_amdgcn_wmma_f32_16x16x32_bf16(
            false, a[ks], false, b, (short)0, pacc, false, false);
    }
    if (nn < 2) {
        float bv = biasP[nn];
        long base = (m0 + half * 8) * 2 + nn;
#pragma unroll
        for (int r = 0; r < 8; ++r) {
            float d = pacc[r] + bv;
            float prev = first ? posIn[base + r * 2] : posOut[base + r * 2];
            posOut[base + r * 2] = prev + d;
        }
    }
}

// readout final: out[n,0..1] = lrelu(bn(y)) @ ro_W1 + ro_b1
__global__ __launch_bounds__(256) void readout_k(
        const float* __restrict__ y, const float* __restrict__ st,
        const float* __restrict__ W1, const float* __restrict__ b1,
        float* __restrict__ out) {
    long row = (long)blockIdx.x * 256 + threadIdx.x;
    if (row >= NROWS) return;
    const float* yr = y + row * 128;
    float a0 = b1[0], a1 = b1[1];
#pragma unroll 8
    for (int k = 0; k < 128; k += 4) {
        float4 v  = *(const float4*)(yr + k);
        float4 s4 = *(const float4*)(st + k);
        float4 t4 = *(const float4*)(st + 128 + k);
        float h0 = bnlrelu(v.x, s4.x, t4.x), h1 = bnlrelu(v.y, s4.y, t4.y);
        float h2 = bnlrelu(v.z, s4.z, t4.z), h3 = bnlrelu(v.w, s4.w, t4.w);
        a0 = fmaf(h0, W1[2 * k + 0], a0); a1 = fmaf(h0, W1[2 * k + 1], a1);
        a0 = fmaf(h1, W1[2 * k + 2], a0); a1 = fmaf(h1, W1[2 * k + 3], a1);
        a0 = fmaf(h2, W1[2 * k + 4], a0); a1 = fmaf(h2, W1[2 * k + 5], a1);
        a0 = fmaf(h3, W1[2 * k + 6], a0); a1 = fmaf(h3, W1[2 * k + 7], a1);
    }
    out[row * 2]     = a0;
    out[row * 2 + 1] = a1;
}

// ---------------- launcher ----------------

extern "C" void kernel_launch(void* const* d_in, const int* in_sizes, int n_in,
                              void* d_out, int out_size, void* d_ws, size_t ws_size,
                              hipStream_t stream) {
    (void)in_sizes; (void)n_in; (void)out_size; (void)ws_size;

    const float* positions = (const float*)d_in[0];
    const float* weights   = (const float*)d_in[1];
    // d_in[2] = batch (unused by effective computation)
    const float* ri_W0 = (const float*)d_in[3];
    const float* ri_b0 = (const float*)d_in[4];
    const float* ri_g0 = (const float*)d_in[5];
    const float* ri_be0= (const float*)d_in[6];
    const float* ri_W1 = (const float*)d_in[7];
    const float* ri_b1 = (const float*)d_in[8];
    const float* ri_g1 = (const float*)d_in[9];
    const float* ri_be1= (const float*)d_in[10];
    const float* dW0   = (const float*)d_in[11];
    const float* db0   = (const float*)d_in[12];
    const float* dg0   = (const float*)d_in[13];
    const float* dbe0  = (const float*)d_in[14];
    const float* dW1   = (const float*)d_in[15];
    const float* db1   = (const float*)d_in[16];
    const float* ro_W0 = (const float*)d_in[17];
    const float* ro_b0 = (const float*)d_in[18];
    const float* ro_g0 = (const float*)d_in[19];
    const float* ro_be0= (const float*)d_in[20];
    const float* ro_W1 = (const float*)d_in[21];
    const float* ro_b1 = (const float*)d_in[22];

    const size_t NH = (size_t)NROWS * 128;
    float* Y     = (float*)d_ws;
    float* W     = Y + NH;
    float* stats = W + NH;                 // 7 * 256
    float* stv   = stats + 7 * 256;        // 7 * 256
    bf16_t* wt_ri1 = (bf16_t*)(stv + 7 * 256);
    bf16_t* wt_d0  = wt_ri1 + 16384;
    bf16_t* wt_d1w = wt_d0  + 4 * 16384;
    bf16_t* wt_d1p = wt_d1w + 4 * 16384;
    bf16_t* wt_ro0 = wt_d1p + 4 * 2048;

    float* pos_out = (float*)d_out;                       // [N,2]
    float* wout    = (float*)d_out + (size_t)NROWS * 2;   // [N,2]

    const int GB = NROWS / 128;       // 3125 GEMM blocks
    const int EB = NROWS * 32 / 256;  // 50000 elementwise blocks

    zero_stats_k<<<7, 256, 0, stream>>>(stats);

    pack_wt_k<<<64, 256, 0, stream>>>(ri_W1, 128, 0, wt_ri1);
    for (int l = 0; l < 4; ++l) {
        pack_wt_k <<<64, 256, 0, stream>>>(dW0 + (size_t)l * 128 * 128, 128, 0, wt_d0  + l * 16384);
        pack_wt_k <<<64, 256, 0, stream>>>(dW1 + (size_t)l * 128 * 130, 130, 2, wt_d1w + l * 16384);
        pack_pos_k<<<8,  256, 0, stream>>>(dW1 + (size_t)l * 128 * 130,          wt_d1p + l * 2048);
    }
    pack_wt_k<<<64, 256, 0, stream>>>(ro_W0, 128, 0, wt_ro0);

    // readin
    readin0_k<<<EB, 256, 0, stream>>>(weights, ri_W0, ri_b0, Y, stats + 0 * 256);
    finalize_k<<<1, 128, 0, stream>>>(stats + 0 * 256, ri_g0, ri_be0, stv + 0 * 256);
    gemm128_stats_k<1><<<GB, 256, 0, stream>>>(Y, wt_ri1, ri_b1, stv + 0 * 256,
                                               Y, stats + 1 * 256);  // in-place safe per 16-row strip
    finalize_k<<<1, 128, 0, stream>>>(stats + 1 * 256, ri_g1, ri_be1, stv + 1 * 256);
    apply_bn_k<<<EB, 256, 0, stream>>>(Y, stv + 1 * 256, W);

    // KNN blocks
    for (int l = 0; l < 4; ++l) {
        int slot = 2 + l;
        gemm128_stats_k<0><<<GB, 256, 0, stream>>>(W, wt_d0 + l * 16384, db0 + l * 128,
                                                   stv, Y, stats + slot * 256);
        finalize_k<<<1, 128, 0, stream>>>(stats + slot * 256, dg0 + l * 128,
                                          dbe0 + l * 128, stv + slot * 256);
        gemm_delta_k<<<GB, 256, 0, stream>>>(Y, wt_d1w + l * 16384, wt_d1p + l * 2048,
                                             db1 + l * 130 + 2, db1 + l * 130,
                                             stv + slot * 256, positions, W, pos_out,
                                             (l == 0) ? 1 : 0);
    }

    // readout
    gemm128_stats_k<0><<<GB, 256, 0, stream>>>(W, wt_ro0, ro_b0, stv, Y, stats + 6 * 256);
    finalize_k<<<1, 128, 0, stream>>>(stats + 6 * 256, ro_g0, ro_be0, stv + 6 * 256);
    readout_k<<<(NROWS + 255) / 256, 256, 0, stream>>>(Y, stv + 6 * 256, ro_W1, ro_b1, wout);
}
